// GATv2Encoder_48223892800362
// MI455X (gfx1250) — matrix operand order repaired and checked by reading, not compiled, not run on hardware
//
#include <hip/hip_runtime.h>
#include <hip/hip_bf16.h>

// ---------------------------------------------------------------------------
// GATv2 encoder (3 layers) for MI455X / gfx1250.
//  - Node transforms (h @ Wl, h @ Wr): LDS-staged B, register-resident A
//    fragments, v_wmma_f32_16x16x32_f16 with f32 accumulation.
//  - Edge phase (dominant, ~8 GB gather/atomic traffic @ 23.3 TB/s HBM):
//    wave-per-edge, float4 coalesced gathers, shfl_xor head reductions,
//    3-pass segment softmax with f32 global atomics.
// ---------------------------------------------------------------------------

#define NEG_SLOPE 0.2f

typedef __attribute__((ext_vector_type(8)))  _Float16 v8h;
typedef __attribute__((ext_vector_type(16))) _Float16 v16h;
typedef __attribute__((ext_vector_type(8)))  float    v8f;

// ---- float atomic max via ordered int trick (init must be -inf) -----------
__device__ __forceinline__ void atomicMaxF32(float* addr, float val) {
  val += 0.0f;  // canonicalize -0 -> +0
  if (val >= 0.0f) atomicMax((int*)addr, __float_as_int(val));
  else             atomicMin((unsigned int*)addr, __float_as_uint(val));
}

// ---------------------------------------------------------------------------
__global__ void fill_f32_kernel(float* __restrict__ p, float v, long long n) {
  long long i = (long long)blockIdx.x * blockDim.x + threadIdx.x;
  long long stride = (long long)gridDim.x * blockDim.x;
  for (; i < n; i += stride) p[i] = v;
}

__global__ void cvt_f32_f16_kernel(const float* __restrict__ in,
                                   _Float16* __restrict__ out, long long n) {
  long long i = (long long)blockIdx.x * blockDim.x + threadIdx.x;
  long long stride = (long long)gridDim.x * blockDim.x;
  for (; i < n; i += stride) out[i] = (_Float16)in[i];
}

// ---------------------------------------------------------------------------
// C[M x NCOLS] = A[M x 128] * B[128 x NCOLS], f16 in / f32 out.
// Block = 256 threads = 8 waves; wave w owns 16-row strip m0=(bx*8+w)*16 and
// sweeps every 16-col tile, so each A fragment set (4x v16h, kept in VGPRs)
// is reused NCOLS/16 times. B is staged in LDS (row pad +8 f16 keeps 16B
// alignment and spreads banks).
// WMMA operand layouts per CDNA5 ISA:
//  A (16-bit 16x32): half=lane>>4: dw0-3 K=kb+8*half..+7, dw4-7 K=kb+16+8*half..
//  B (16-bit 32x16): lane L holds col N=n0+(L&15), K same as A (kb+8*half.., kb+16+8*half..).
//  C/D: VGPR r -> row m0 + 8*half + r, col n0 + (lane&15).
template <int NCOLS>
__global__ void gemm_f16_wmma_kernel(const _Float16* __restrict__ A,
                                     const _Float16* __restrict__ Bw,
                                     float* __restrict__ C, int M) {
  constexpr int K   = 128;
  constexpr int LDB = NCOLS + 8;                    // f16 units; %8==0 -> 16B ok
  __shared__ _Float16 Bs[K * (128 + 8)];            // sized for worst case

  // ---- cooperative B stage: global -> LDS (8 f16 = 16B per thread/iter) ----
  constexpr int TOTB = K * NCOLS;
  for (int base = threadIdx.x * 8; base < TOTB; base += 256 * 8) {
    const int krow = base / NCOLS;
    const int col  = base - krow * NCOLS;
    *(v8h*)(&Bs[krow * LDB + col]) = *(const v8h*)(Bw + base);
  }
  __syncthreads();

  const int lane = threadIdx.x & 31;
  const int half = lane >> 4;
  const int m0 = (blockIdx.x * 8 + (threadIdx.x >> 5)) * 16;
  if (m0 >= M) return;                              // after barrier: safe

  long long mrow = m0 + (lane & 15);
  if (mrow >= M) mrow = M - 1;                      // clamp (tail tiles only)
  const _Float16* arow = A + mrow * (long long)K;

  v16h af[4];
#pragma unroll
  for (int kk = 0; kk < 4; ++kk) {
    v8h alo = *(const v8h*)(arow + kk * 32 + half * 8);
    v8h ahi = *(const v8h*)(arow + kk * 32 + 16 + half * 8);
#pragma unroll
    for (int i = 0; i < 8; ++i) { af[kk][i] = alo[i]; af[kk][8 + i] = ahi[i]; }
  }

  const bool fullTile = (m0 + 16 <= M);
#pragma unroll
  for (int n0 = 0; n0 < NCOLS; n0 += 16) {
    v8f acc = {};
#pragma unroll
    for (int kk = 0; kk < 4; ++kk) {
      const _Float16* bcol = &Bs[(kk * 32 + half * 8) * LDB + n0 + (lane & 15)];
      v16h bf;
#pragma unroll
      for (int i = 0; i < 8; ++i) { bf[i] = bcol[i * LDB]; bf[8 + i] = bcol[(16 + i) * LDB]; }
      acc = __builtin_amdgcn_wmma_f32_16x16x32_f16(
          /*neg_a=*/false, af[kk], /*neg_b=*/false, bf,
          /*c_mod=*/(short)0, acc, /*reuse_a=*/false, /*reuse_b=*/false);
    }
    float* crow = C + (long long)(m0 + half * 8) * NCOLS + n0 + (lane & 15);
    if (fullTile) {
#pragma unroll
      for (int r = 0; r < 8; ++r) crow[(long long)r * NCOLS] = acc[r];
    } else {
#pragma unroll
      for (int r = 0; r < 8; ++r)
        if (m0 + half * 8 + r < M) crow[(long long)r * NCOLS] = acc[r];
    }
  }
}

// ---------------------------------------------------------------------------
// Pass A: per-edge attention logit + segment max.  Wave per edge.
__global__ void edge_score_kernel(const long long* __restrict__ ei,
                                  long long E, long long Etot,
                                  const float* __restrict__ hl,
                                  const float* __restrict__ hr,
                                  const float* __restrict__ att,
                                  float* __restrict__ score,
                                  float* __restrict__ smax,
                                  int TOT, int CH, int H) {
  const long long e = (long long)blockIdx.x * (blockDim.x >> 5) + (threadIdx.x >> 5);
  if (e >= Etot) return;
  const int lane = threadIdx.x & 31;
  long long s, d;
  if (e < E) { s = ei[e]; d = ei[E + e]; } else { s = d = e - E; }

  const int g = lane * 4;
  float part = 0.0f;
  if (g < TOT) {
    const float4 va = *(const float4*)(hl + s * (long long)TOT + g);
    const float4 vb = *(const float4*)(hr + d * (long long)TOT + g);
    const float4 aa = *(const float4*)(att + g);
    float t;
    t = va.x + vb.x; part += aa.x * (t > 0.0f ? t : NEG_SLOPE * t);
    t = va.y + vb.y; part += aa.y * (t > 0.0f ? t : NEG_SLOPE * t);
    t = va.z + vb.z; part += aa.z * (t > 0.0f ? t : NEG_SLOPE * t);
    t = va.w + vb.w; part += aa.w * (t > 0.0f ? t : NEG_SLOPE * t);
  }
  const int lanesPerHead = CH >> 2;  // 8 (CH=32) or 16 (CH=64)
  for (int off = 1; off < lanesPerHead; off <<= 1)
    part += __shfl_xor(part, off, 32);
  if (g < TOT && (lane & (lanesPerHead - 1)) == 0) {
    const int head = g / CH;
    score[e * H + head] = part;
    atomicMaxF32(&smax[d * H + head], part);
  }
}

// Pass B: w = exp(score - smax[dst]); denom[dst] += w.  Thread per (edge,head).
__global__ void edge_exp_kernel(const long long* __restrict__ ei,
                                long long E, long long Etot,
                                float* __restrict__ score,
                                const float* __restrict__ smax,
                                float* __restrict__ denom, int H) {
  const long long total = Etot * H;
  long long idx = (long long)blockIdx.x * blockDim.x + threadIdx.x;
  long long stride = (long long)gridDim.x * blockDim.x;
  for (; idx < total; idx += stride) {
    const long long e = idx / H;
    const int head = (int)(idx - e * H);
    const long long d = (e < E) ? ei[E + e] : (e - E);
    const float w = __expf(score[idx] - smax[d * H + head]);
    score[idx] = w;
    atomicAdd(&denom[d * H + head], w);
  }
}

// Pass C: acc[dst] += hl[src] * alpha.  Wave per edge, float4 per lane.
__global__ void edge_aggr_kernel(const long long* __restrict__ ei,
                                 long long E, long long Etot,
                                 const float* __restrict__ hl,
                                 const float* __restrict__ score,
                                 const float* __restrict__ denom,
                                 float* __restrict__ acc,
                                 int TOT, int CH, int H) {
  const long long e = (long long)blockIdx.x * (blockDim.x >> 5) + (threadIdx.x >> 5);
  if (e >= Etot) return;
  const int lane = threadIdx.x & 31;
  const int g = lane * 4;
  if (g >= TOT) return;
  long long s, d;
  if (e < E) { s = ei[e]; d = ei[E + e]; } else { s = d = e - E; }
  const int head = g / CH;
  const float alpha = score[e * H + head] / (denom[d * H + head] + 1e-16f);
  const float4 v = *(const float4*)(hl + s * (long long)TOT + g);
  float* base = acc + d * (long long)TOT + g;
  atomicAdd(base + 0, v.x * alpha);
  atomicAdd(base + 1, v.y * alpha);
  atomicAdd(base + 2, v.z * alpha);
  atomicAdd(base + 3, v.w * alpha);
}

// Finalize: relu(acc + bias) -> f16 activation buffer (layers 0,1) or f32 d_out.
__global__ void finalize_kernel(const float* __restrict__ acc,
                                const float* __restrict__ bias,
                                _Float16* __restrict__ h16,
                                float* __restrict__ f32out,
                                int TOT, long long total) {
  long long idx = (long long)blockIdx.x * blockDim.x + threadIdx.x;
  long long stride = (long long)gridDim.x * blockDim.x;
  for (; idx < total; idx += stride) {
    const int c = (int)(idx % TOT);
    float v = acc[idx] + bias[c];
    v = v > 0.0f ? v : 0.0f;
    if (h16) h16[idx] = (_Float16)v;
    else     f32out[idx] = v;
  }
}

// ---------------------------------------------------------------------------
static inline size_t align256(size_t x) { return (x + 255) & ~(size_t)255; }

extern "C" void kernel_launch(void* const* d_in, const int* in_sizes, int n_in,
                              void* d_out, int out_size, void* d_ws, size_t ws_size,
                              hipStream_t stream) {
  const long long N    = in_sizes[0] / 128;
  const long long E    = in_sizes[1] / 2;
  const long long Etot = E + N;
  const int K = 128;

  const float*     x  = (const float*)d_in[0];
  const long long* ei = (const long long*)d_in[1];  // reference dtype int64
  const float* Wl[3]   = {(const float*)d_in[2], (const float*)d_in[6],  (const float*)d_in[10]};
  const float* Wr[3]   = {(const float*)d_in[3], (const float*)d_in[7],  (const float*)d_in[11]};
  const float* att[3]  = {(const float*)d_in[4], (const float*)d_in[8],  (const float*)d_in[12]};
  const float* bias[3] = {(const float*)d_in[5], (const float*)d_in[9],  (const float*)d_in[13]};
  const int TOT[3] = {128, 128, 64};
  const int CH[3]  = {32, 32, 64};
  const int HH[3]  = {4, 4, 1};

  // -------- workspace layout (~210 MB) --------
  char* w = (char*)d_ws;
  size_t off = 0;
  _Float16* h16  = (_Float16*)(w + off); off += align256((size_t)N * 128 * sizeof(_Float16));
  float*    hl   = (float*)   (w + off); off += align256((size_t)N * 128 * sizeof(float));
  float*    hr   = (float*)   (w + off); off += align256((size_t)N * 128 * sizeof(float));
  float*    acc  = (float*)   (w + off); off += align256((size_t)N * 128 * sizeof(float));
  float*    smax = (float*)   (w + off); off += align256((size_t)N * 4   * sizeof(float));
  float*    den  = (float*)   (w + off); off += align256((size_t)N * 4   * sizeof(float));
  float*    scr  = (float*)   (w + off); off += align256((size_t)Etot * 4 * sizeof(float));
  _Float16* w16a = (_Float16*)(w + off); off += align256((size_t)K * 128 * sizeof(_Float16));
  _Float16* w16b = (_Float16*)(w + off); off += align256((size_t)K * 128 * sizeof(_Float16));
  (void)ws_size; (void)n_in; (void)out_size;

  const int TPB = 256;
  const long long nInit = N * 128;

  // initial activation: f32 -> f16
  cvt_f32_f16_kernel<<<(int)((nInit + TPB - 1) / TPB), TPB, 0, stream>>>(x, h16, nInit);

  const int edgeBlocks = (int)((Etot + 7) / 8);          // 8 waves (edges) / block
  const int gemmBlocks = (int)((N + 127) / 128);         // 8 row strips / block

  for (int l = 0; l < 3; ++l) {
    const int tot = TOT[l], ch = CH[l], h = HH[l];
    const long long wsz = (long long)K * tot;

    // weights f32 -> f16
    cvt_f32_f16_kernel<<<(int)((wsz + TPB - 1) / TPB), TPB, 0, stream>>>(Wl[l], w16a, wsz);
    cvt_f32_f16_kernel<<<(int)((wsz + TPB - 1) / TPB), TPB, 0, stream>>>(Wr[l], w16b, wsz);

    // hl = h @ Wl, hr = h @ Wr   (WMMA, LDS-staged B, A fragments in VGPRs)
    if (tot == 128) {
      gemm_f16_wmma_kernel<128><<<gemmBlocks, TPB, 0, stream>>>(h16, w16a, hl, (int)N);
      gemm_f16_wmma_kernel<128><<<gemmBlocks, TPB, 0, stream>>>(h16, w16b, hr, (int)N);
    } else {
      gemm_f16_wmma_kernel<64><<<gemmBlocks, TPB, 0, stream>>>(h16, w16a, hl, (int)N);
      gemm_f16_wmma_kernel<64><<<gemmBlocks, TPB, 0, stream>>>(h16, w16b, hr, (int)N);
    }

    // segment-softmax state
    const long long nAcc = N * tot, nSeg = N * h;
    fill_f32_kernel<<<(int)((nAcc + TPB - 1) / TPB), TPB, 0, stream>>>(acc, 0.0f, nAcc);
    fill_f32_kernel<<<(int)((nSeg + TPB - 1) / TPB), TPB, 0, stream>>>(smax, -__builtin_inff(), nSeg);
    fill_f32_kernel<<<(int)((nSeg + TPB - 1) / TPB), TPB, 0, stream>>>(den, 0.0f, nSeg);

    // edge passes
    edge_score_kernel<<<edgeBlocks, TPB, 0, stream>>>(ei, E, Etot, hl, hr, att[l],
                                                      scr, smax, tot, ch, h);
    const long long nEH = Etot * h;
    edge_exp_kernel<<<(int)((nEH + TPB - 1) / TPB), TPB, 0, stream>>>(ei, E, Etot,
                                                                      scr, smax, den, h);
    edge_aggr_kernel<<<edgeBlocks, TPB, 0, stream>>>(ei, E, Etot, hl, scr, den,
                                                     acc, tot, ch, h);

    // bias + relu -> next-layer f16 activations, or final f32 output
    finalize_kernel<<<(int)((nAcc + TPB - 1) / TPB), TPB, 0, stream>>>(
        acc, bias[l], (l < 2) ? h16 : (_Float16*)nullptr,
        (l < 2) ? (float*)nullptr : (float*)d_out, tot, nAcc);
  }
}
